// SimpleSelfAttention_78426102825675
// MI455X (gfx1250) — matrix operand order repaired and checked by reading, not compiled, not run on hardware
//
#include <hip/hip_runtime.h>
#include <hip/hip_bf16.h>
#include <math.h>

// ---------------------------------------------------------------------------
// Problem constants (from the reference)
// ---------------------------------------------------------------------------
#define D_MODEL 2048
#define BATCH   4
#define SEQ     2048
#define PASTLEN 2048
#define SKV     4096            // PASTLEN + SEQ
#define M_TOK   (BATCH * SEQ)   // 8192 tokens

typedef __attribute__((ext_vector_type(16))) __bf16          v16bf;
typedef __attribute__((ext_vector_type(8)))  float           v8f;
typedef __attribute__((ext_vector_type(8)))  unsigned short  ushort8;
typedef __attribute__((ext_vector_type(16))) unsigned short  ushort16;

__device__ __forceinline__ unsigned short f32_to_bf16_rte(float f) {
    union { float f; unsigned u; } v; v.f = f;
    unsigned r = v.u + 0x7FFFu + ((v.u >> 16) & 1u);   // round-to-nearest-even
    return (unsigned short)(r >> 16);
}

// A-operand fragment (16x32 MxK, bf16): lane m = lane&15, khalf = lane>>4.
// elements 0..7  = A[m][k0 + khalf*8 + 0..7]
// elements 8..15 = A[m][k0 + 16 + khalf*8 + 0..7]
__device__ __forceinline__ v16bf load_a_frag(const unsigned short* __restrict__ A,
                                             size_t row, int ldk, int k0, int khalf) {
    const unsigned short* p = A + row * (size_t)ldk + k0 + khalf * 8;
    union { ushort8 h[2]; ushort16 s; } u;
    u.h[0] = *(const ushort8*)(p);
    u.h[1] = *(const ushort8*)(p + 16);
    return __builtin_bit_cast(v16bf, u.s);
}

// B-operand fragment (32x16 KxN, bf16) for C = A * Brm^T where Brm is row-major
// [N][K]: lane n = lane&15, khalf = lane>>4; elements 0..7 = Brm[n][k0+khalf*8+0..7],
// elements 8..15 = Brm[n][k0+16+khalf*8+0..7].
__device__ __forceinline__ v16bf load_b_frag(const unsigned short* __restrict__ B,
                                             size_t row, int ldk, int k0, int khalf) {
    const unsigned short* p = B + row * (size_t)ldk + k0 + khalf * 8;
    union { ushort8 h[2]; ushort16 s; } u;
    u.h[0] = *(const ushort8*)(p);
    u.h[1] = *(const ushort8*)(p + 16);
    return __builtin_bit_cast(v16bf, u.s);
}

__device__ __forceinline__ v8f wmma_bf16(v16bf a, v16bf b, v8f c) {
    return __builtin_amdgcn_wmma_f32_16x16x32_bf16(false, a, false, b,
                                                   (short)0, c, false, false);
}

// ---------------------------------------------------------------------------
// Elementwise helpers
// ---------------------------------------------------------------------------
__global__ void cvt_f32_to_bf16_kernel(const float* __restrict__ s,
                                       unsigned short* __restrict__ d, size_t n) {
    size_t i = (size_t)blockIdx.x * blockDim.x + threadIdx.x;
    size_t stride = (size_t)gridDim.x * blockDim.x;
    for (; i < n; i += stride) d[i] = f32_to_bf16_rte(s[i]);
}

// past [B][PASTLEN][D] -> cache rows [0, PASTLEN) of [B][SKV][D]
__global__ void copy_past_kernel(const float* __restrict__ src,
                                 float* __restrict__ dst,
                                 size_t n_per_b, size_t dst_batch_stride) {
    size_t total = (size_t)BATCH * n_per_b;
    size_t i = (size_t)blockIdx.x * blockDim.x + threadIdx.x;
    size_t stride = (size_t)gridDim.x * blockDim.x;
    for (; i < total; i += stride) {
        size_t b = i / n_per_b, r = i - b * n_per_b;
        dst[b * dst_batch_stride + r] = src[i];
    }
}

// v_cache fp32 [B][SKV][D]  ->  V^T bf16 [B][D][SKV]  (LDS-tiled transpose)
__global__ __launch_bounds__(256)
void transpose_cvt_kernel(const float* __restrict__ src,
                          unsigned short* __restrict__ dst) {
    __shared__ float tile[32][33];
    const int rows = SKV, cols = D_MODEL;
    const int b  = blockIdx.z;
    const int c0 = blockIdx.x * 32;
    const int r0 = blockIdx.y * 32;
    const float* s = src + (size_t)b * rows * cols;
    unsigned short* d = dst + (size_t)b * rows * cols;
    for (int i = threadIdx.y; i < 32; i += 8)
        tile[i][threadIdx.x] = s[(size_t)(r0 + i) * cols + c0 + threadIdx.x];
    __syncthreads();
    for (int i = threadIdx.y; i < 32; i += 8)
        d[(size_t)(c0 + i) * rows + r0 + threadIdx.x] =
            f32_to_bf16_rte(tile[threadIdx.x][i]);
}

// ---------------------------------------------------------------------------
// WMMA GEMM:  C[m][n] = sum_k A[m][k] * Brm[n][k] + bias[n]
// 256 threads = 8 waves (2x4), block tile 128x256, wave tile 64x64 (4x4 frags).
// Output rows remapped for KV-cache writes:
//   dst_row(m) = (m / SEQ) * ld_batch + row_off + (m % SEQ)
// ---------------------------------------------------------------------------
__global__ __launch_bounds__(256)
void gemm_bf16_wmma(const unsigned short* __restrict__ A,   // [M][K] bf16
                    const unsigned short* __restrict__ Bm,  // [N][K] bf16
                    const float* __restrict__ bias,         // [N]
                    float* __restrict__ Cf,                 // optional fp32 out
                    unsigned short* __restrict__ Cb,        // optional bf16 out
                    int N, int K, int ld_batch, int row_off) {
    const int lane  = threadIdx.x & 31;
    const int wave  = threadIdx.x >> 5;
    const int waveM = wave >> 2, waveN = wave & 3;
    const int n16   = lane & 15, half = lane >> 4;
    const size_t rowW = (size_t)blockIdx.y * 128 + waveM * 64;
    const size_t colW = (size_t)blockIdx.x * 256 + waveN * 64;

    v8f acc[4][4];
    #pragma unroll
    for (int i = 0; i < 4; ++i)
        #pragma unroll
        for (int j = 0; j < 4; ++j)
            #pragma unroll
            for (int e = 0; e < 8; ++e) acc[i][j][e] = 0.0f;

    for (int k0 = 0; k0 < K; k0 += 32) {
        if (k0 + 128 < K) {   // hint next tiles into cache (global_prefetch_b8)
            __builtin_prefetch(A  + (rowW + n16) * (size_t)K + k0 + 128, 0, 1);
            __builtin_prefetch(Bm + (colW + n16) * (size_t)K + k0 + 128, 0, 1);
        }
        v16bf a[4], b[4];
        #pragma unroll
        for (int i = 0; i < 4; ++i)
            a[i] = load_a_frag(A, rowW + i * 16 + n16, K, k0, half);
        #pragma unroll
        for (int j = 0; j < 4; ++j)
            b[j] = load_b_frag(Bm, colW + j * 16 + n16, K, k0, half);
        #pragma unroll
        for (int i = 0; i < 4; ++i)
            #pragma unroll
            for (int j = 0; j < 4; ++j)
                acc[i][j] = wmma_bf16(a[i], b[j], acc[i][j]);
    }

    #pragma unroll
    for (int i = 0; i < 4; ++i) {
        #pragma unroll
        for (int r = 0; r < 8; ++r) {
            size_t m    = rowW + i * 16 + half * 8 + r;
            size_t drow = (m / SEQ) * (size_t)ld_batch + row_off + (m % SEQ);
            #pragma unroll
            for (int j = 0; j < 4; ++j) {
                size_t n = colW + j * 16 + n16;
                float  v = acc[i][j][r] + bias[n];
                if (Cf) Cf[drow * (size_t)N + n] = v;
                if (Cb) Cb[drow * (size_t)N + n] = f32_to_bf16_rte(v);
            }
        }
    }
}

// ---------------------------------------------------------------------------
// Attention: one block per (batch, 16-query tile). 8 waves.
// Phase 1: scores[16][SKV] = (Q_tile · Kc^T) * scale   -> fp32 LDS (256 KB)
// Phase 2: fp32 softmax across rows (in LDS)
// Phase 3: attn = P · V via WMMA, P read from LDS (cvt to bf16), V^T from global
// ---------------------------------------------------------------------------
__global__ __launch_bounds__(256)
void attention_wmma(const unsigned short* __restrict__ Q,    // [B][SEQ][D] bf16
                    const unsigned short* __restrict__ Kc,   // [B][SKV][D] bf16
                    const unsigned short* __restrict__ VT,   // [B][D][SKV] bf16
                    unsigned short* __restrict__ Oattn) {    // [B][SEQ][D] bf16
    __shared__ float sc[16 * SKV];   // 256 KB of the 320 KB WGP LDS
    __shared__ float red[256];
    __shared__ float rstat[16];

    const int lane = threadIdx.x & 31;
    const int wave = threadIdx.x >> 5;
    const int n16  = lane & 15, half = lane >> 4;
    const int b    = blockIdx.x >> 7;          // SEQ/16 = 128 q-tiles per batch
    const int qt   = blockIdx.x & 127;
    const size_t qrow0 = (size_t)b * SEQ + qt * 16;
    const size_t krow0 = (size_t)b * SKV;
    const float  scale = 0.022097086912079608f;   // 1/sqrt(2048)

    // ---- Phase 1: scores -> LDS. 256 key-tiles, 8 waves x 8 groups x 4 tiles.
    for (int g = 0; g < 8; ++g) {
        const int nt0 = (wave * 8 + g) * 4;
        v8f acc[4];
        #pragma unroll
        for (int j = 0; j < 4; ++j)
            #pragma unroll
            for (int e = 0; e < 8; ++e) acc[j][e] = 0.0f;

        for (int k0 = 0; k0 < D_MODEL; k0 += 32) {
            v16bf a = load_a_frag(Q, qrow0 + n16, D_MODEL, k0, half);
            #pragma unroll
            for (int j = 0; j < 4; ++j) {
                v16bf bf = load_b_frag(Kc, krow0 + (size_t)(nt0 + j) * 16 + n16,
                                       D_MODEL, k0, half);
                acc[j] = wmma_bf16(a, bf, acc[j]);
            }
        }
        #pragma unroll
        for (int j = 0; j < 4; ++j)
            #pragma unroll
            for (int r = 0; r < 8; ++r)
                sc[(half * 8 + r) * SKV + (nt0 + j) * 16 + n16] = acc[j][r] * scale;
    }
    __syncthreads();

    // ---- Phase 2: softmax, 16 threads per row.
    {
        const int row = threadIdx.x >> 4, sub = threadIdx.x & 15;
        float* prow = sc + row * SKV;
        float mx = -3.4e38f;
        for (int c = sub; c < SKV; c += 16) mx = fmaxf(mx, prow[c]);
        red[threadIdx.x] = mx;
        __syncthreads();
        if (sub == 0) {
            float m2 = red[row * 16];
            for (int t = 1; t < 16; ++t) m2 = fmaxf(m2, red[row * 16 + t]);
            rstat[row] = m2;
        }
        __syncthreads();
        const float m2 = rstat[row];
        float s = 0.0f;
        for (int c = sub; c < SKV; c += 16) {
            float e = __expf(prow[c] - m2);
            prow[c] = e;
            s += e;
        }
        red[threadIdx.x] = s;
        __syncthreads();
        if (sub == 0) {
            float t2 = 0.0f;
            for (int t = 0; t < 16; ++t) t2 += red[row * 16 + t];
            rstat[row] = 1.0f / t2;
        }
        __syncthreads();
        const float inv = rstat[row];
        for (int c = sub; c < SKV; c += 16) prow[c] *= inv;
    }
    __syncthreads();

    // ---- Phase 3: attn = P · V. 128 d-tiles, 8 waves x 4 groups x 4 tiles.
    for (int g = 0; g < 4; ++g) {
        const int dt0 = (wave * 4 + g) * 4;
        v8f acc[4];
        #pragma unroll
        for (int j = 0; j < 4; ++j)
            #pragma unroll
            for (int e = 0; e < 8; ++e) acc[j][e] = 0.0f;

        for (int kc = 0; kc < SKV; kc += 32) {
            // A fragment from LDS probabilities, converted fp32 -> bf16.
            const float* p = sc + n16 * SKV + kc + half * 8;
            ushort16 u;
            #pragma unroll
            for (int i = 0; i < 8; ++i) u[i]     = f32_to_bf16_rte(p[i]);
            #pragma unroll
            for (int i = 0; i < 8; ++i) u[8 + i] = f32_to_bf16_rte(p[16 + i]);
            v16bf a = __builtin_bit_cast(v16bf, u);
            #pragma unroll
            for (int j = 0; j < 4; ++j) {
                v16bf bf = load_b_frag(VT,
                    (size_t)b * D_MODEL + (size_t)(dt0 + j) * 16 + n16,
                    SKV, kc, half);
                acc[j] = wmma_bf16(a, bf, acc[j]);
            }
        }
        #pragma unroll
        for (int j = 0; j < 4; ++j)
            #pragma unroll
            for (int r = 0; r < 8; ++r) {
                size_t m = qrow0 + half * 8 + r;
                Oattn[m * D_MODEL + (dt0 + j) * 16 + n16] =
                    f32_to_bf16_rte(acc[j][r]);
            }
    }
}

// ---------------------------------------------------------------------------
// Host launcher
// ---------------------------------------------------------------------------
extern "C" void kernel_launch(void* const* d_in, const int* in_sizes, int n_in,
                              void* d_out, int out_size, void* d_ws, size_t ws_size,
                              hipStream_t stream) {
    (void)in_sizes; (void)n_in; (void)out_size; (void)ws_size;

    const float* x      = (const float*)d_in[0];
    const float* past_k = (const float*)d_in[1];
    const float* past_v = (const float*)d_in[2];
    const float* Wq     = (const float*)d_in[3];
    const float* bq     = (const float*)d_in[4];
    const float* Wk     = (const float*)d_in[5];
    const float* bk     = (const float*)d_in[6];
    const float* Wv     = (const float*)d_in[7];
    const float* bv     = (const float*)d_in[8];
    const float* Wo     = (const float*)d_in[9];
    const float* bo     = (const float*)d_in[10];

    // d_out = [ out | k_cache | v_cache ] (tuple flattening order)
    float* out = (float*)d_out;                                 // [B][SEQ][D]
    float* kc  = out + (size_t)BATCH * SEQ * D_MODEL;           // [B][SKV][D]
    float* vc  = kc  + (size_t)BATCH * SKV * D_MODEL;           // [B][SKV][D]

    // Workspace layout (224 MB). attn bf16 aliases x bf16 (x dead by then).
    char* ws = (char*)d_ws;
    unsigned short* xbf  = (unsigned short*)(ws);                    // 32 MB
    unsigned short* qbf  = (unsigned short*)(ws + (size_t)33554432); // 32 MB
    unsigned short* kbf  = (unsigned short*)(ws + (size_t)67108864); // 64 MB
    unsigned short* vTbf = (unsigned short*)(ws + (size_t)134217728);// 64 MB
    unsigned short* wqbf = (unsigned short*)(ws + (size_t)201326592);// 4x8 MB
    unsigned short* wkbf = wqbf + (size_t)D_MODEL * D_MODEL;
    unsigned short* wvbf = wkbf + (size_t)D_MODEL * D_MODEL;
    unsigned short* wobf = wvbf + (size_t)D_MODEL * D_MODEL;
    unsigned short* attnbf = xbf;   // alias

    const size_t nTok = (size_t)M_TOK * D_MODEL;      // 16.7M
    const size_t nW   = (size_t)D_MODEL * D_MODEL;    // 4.2M
    const size_t nKV  = (size_t)BATCH * SKV * D_MODEL;// 33.5M

    // 1) fp32 -> bf16 conversions of activations and weights
    cvt_f32_to_bf16_kernel<<<4096, 256, 0, stream>>>(x,  xbf,  nTok);
    cvt_f32_to_bf16_kernel<<<1024, 256, 0, stream>>>(Wq, wqbf, nW);
    cvt_f32_to_bf16_kernel<<<1024, 256, 0, stream>>>(Wk, wkbf, nW);
    cvt_f32_to_bf16_kernel<<<1024, 256, 0, stream>>>(Wv, wvbf, nW);
    cvt_f32_to_bf16_kernel<<<1024, 256, 0, stream>>>(Wo, wobf, nW);

    // 2) past_k/past_v -> front of fp32 caches in d_out
    copy_past_kernel<<<4096, 256, 0, stream>>>(past_k, kc,
        (size_t)PASTLEN * D_MODEL, (size_t)SKV * D_MODEL);
    copy_past_kernel<<<4096, 256, 0, stream>>>(past_v, vc,
        (size_t)PASTLEN * D_MODEL, (size_t)SKV * D_MODEL);

    // 3) QKV projections (WMMA).  grid: N/256 x M/128 = 8 x 64
    dim3 ggrid(D_MODEL / 256, M_TOK / 128);
    gemm_bf16_wmma<<<ggrid, 256, 0, stream>>>(xbf, wqbf, bq,
        nullptr, qbf, D_MODEL, D_MODEL, SEQ, 0);               // q -> bf16 flat
    gemm_bf16_wmma<<<ggrid, 256, 0, stream>>>(xbf, wkbf, bk,
        kc, nullptr, D_MODEL, D_MODEL, SKV, PASTLEN);          // k -> cache fp32
    gemm_bf16_wmma<<<ggrid, 256, 0, stream>>>(xbf, wvbf, bv,
        vc, nullptr, D_MODEL, D_MODEL, SKV, PASTLEN);          // v -> cache fp32

    // 4) full caches -> bf16 (K row-major, V transposed [B][D][SKV])
    cvt_f32_to_bf16_kernel<<<4096, 256, 0, stream>>>(kc, kbf, nKV);
    {
        dim3 tgrid(D_MODEL / 32, SKV / 32, BATCH);
        dim3 tblk(32, 8);
        transpose_cvt_kernel<<<tgrid, tblk, 0, stream>>>(vc, vTbf);
    }

    // 5) attention: one block per (batch, 16-query tile)
    attention_wmma<<<BATCH * (SEQ / 16), 256, 0, stream>>>(qbf, kbf, vTbf, attnbf);

    // 6) output projection -> fp32 out
    gemm_bf16_wmma<<<ggrid, 256, 0, stream>>>(attnbf, wobf, bo,
        out, nullptr, D_MODEL, D_MODEL, SEQ, 0);
}
